// MambaFinetune_33105607917975
// MI455X (gfx1250) — compile-verified, matrix operand order checked by reading, not run on hardware
//
#include <hip/hip_runtime.h>

// ---------------------------------------------------------------------------
// MI455X (gfx1250) Mamba-ECG forward.
// GEMMs: v_wmma_f32_16x16x32_bf16 with GLOBAL_LOAD_ASYNC_TO_LDS_B128 staging
// (ASYNCcnt-tracked DMA, single barrier per K-step).
// ---------------------------------------------------------------------------

typedef __bf16 bf16_t;
typedef __attribute__((ext_vector_type(16))) __bf16 v16bf;
typedef __attribute__((ext_vector_type(8)))  float  v8f;

#define D_MODEL  512
#define D_STATE  16
#define D_INNER  1024
#define DT_RANK  32
#define D_CONV   4
#define N_LAYERS 4
#define NUM_LEADS 12
#define BATCH    8
#define SEQLEN   2500
#define ML       (BATCH * SEQLEN)   // 20000 rows

// ---------------------------------------------------------------------------
// bf16 WMMA GEMM:  C[M,N] = A[M,K] * B[N,K]^T   (A row-major, B = weight [N,K])
// Block tile 128x128, K-step 32, 8 waves (2x4), each wave 64x32 = 4x2 WMMA tiles.
// ---------------------------------------------------------------------------
#define BM 128
#define BN 128
#define BK 32
#define LDSS 48   // LDS row stride in bf16 (96 B: keeps 32B fragment alignment)

// Async DMA one 128x32 bf16 tile into LDS. Rows are clamped (not zero-filled):
// out-of-range A rows / B rows only feed predicated-out outputs.
__device__ __forceinline__ void async_tile_load(
    const bf16_t* __restrict__ src, int rowBase, int numRows, int ldk, int k0,
    bf16_t* dstLds, int tid)
{
  const int r0 = tid >> 2;            // 0..63
  const int c  = (tid & 3) * 8;       // 0,8,16,24 (bf16)
#pragma unroll
  for (int p = 0; p < 2; ++p) {
    const int r  = r0 + 64 * p;
    int gr = rowBase + r;
    gr = gr < numRows ? gr : numRows - 1;
    const bf16_t* g = src + (size_t)gr * ldk + k0 + c;
    const unsigned lds = (unsigned)(uintptr_t)(dstLds + r * LDSS + c);
    asm volatile("global_load_async_to_lds_b128 %0, %1, off"
                 :: "v"(lds), "v"(g) : "memory");
  }
}

__device__ __forceinline__ void wait_async_all()
{
  asm volatile("s_wait_asynccnt 0x0" ::: "memory");
}

__global__ __launch_bounds__(256)
void gemm_bf16_wmma_kernel(const bf16_t* __restrict__ A,
                           const bf16_t* __restrict__ B,
                           float* __restrict__ C,
                           int M, int N, int K)
{
  __shared__ __align__(32) bf16_t As[2][BM * LDSS];
  __shared__ __align__(32) bf16_t Bs[2][BN * LDSS];

  const int tid  = threadIdx.x;
  const int lane = tid & 31;
  const int wave = tid >> 5;
  const int wm   = wave & 1;    // 0..1 -> 64 rows each
  const int wn   = wave >> 1;   // 0..3 -> 32 cols each
  const int bm0  = blockIdx.y * BM;
  const int bn0  = blockIdx.x * BN;

  v8f acc[4][2];
#pragma unroll
  for (int i = 0; i < 4; ++i)
#pragma unroll
    for (int j = 0; j < 2; ++j)
      acc[i][j] = (v8f){0.f,0.f,0.f,0.f,0.f,0.f,0.f,0.f};

  const int lrow = lane & 15;
  const int lko  = (lane >> 4) * 16;

  // Prologue: DMA first K-tile into buffer 0.
  async_tile_load(A, bm0, M, K, 0, As[0], tid);
  async_tile_load(B, bn0, N, K, 0, Bs[0], tid);

  const int nk = K / BK;
  for (int kt = 0; kt < nk; ++kt) {
    const int cur = kt & 1;

    // Current tile's async writes landed (per-wave), then cross-wave fence.
    // The same barrier also guarantees every wave is done READING buffer
    // cur^1 (consumed last iteration), so we may DMA over it below.
    wait_async_all();
    __syncthreads();

    if (kt + 1 < nk) {
      async_tile_load(A, bm0, M, K, (kt + 1) * BK, As[cur ^ 1], tid);
      async_tile_load(B, bn0, N, K, (kt + 1) * BK, Bs[cur ^ 1], tid);
    }

    v16bf afr[4], bfr[2];
#pragma unroll
    for (int i = 0; i < 4; ++i) {
      const int r = wm * 64 + i * 16 + lrow;
      afr[i] = *reinterpret_cast<const v16bf*>(&As[cur][r * LDSS + lko]);
    }
#pragma unroll
    for (int j = 0; j < 2; ++j) {
      const int r = wn * 32 + j * 16 + lrow;
      bfr[j] = *reinterpret_cast<const v16bf*>(&Bs[cur][r * LDSS + lko]);
    }
#pragma unroll
    for (int i = 0; i < 4; ++i)
#pragma unroll
      for (int j = 0; j < 2; ++j)
        acc[i][j] = __builtin_amdgcn_wmma_f32_16x16x32_bf16(
            false, afr[i], false, bfr[j], (short)0, acc[i][j], false, false);
  }

  // Epilogue: C/D layout — lane<16: rows v+0..7, lane>=16: rows v+8..15; col = lane%16
#pragma unroll
  for (int i = 0; i < 4; ++i) {
    const int rbase = bm0 + wm * 64 + i * 16 + (lane >> 4) * 8;
#pragma unroll
    for (int j = 0; j < 2; ++j) {
      const int col = bn0 + wn * 32 + j * 16 + (lane & 15);
      if (col < N) {
#pragma unroll
        for (int v = 0; v < 8; ++v) {
          const int r = rbase + v;
          if (r < M) C[(size_t)r * N + col] = acc[i][j][v];
        }
      }
    }
  }
}

// ---------------------------------------------------------------------------
// f32 -> bf16 elementwise (weights + activations)
// ---------------------------------------------------------------------------
__global__ void f32_to_bf16_kernel(const float* __restrict__ in,
                                   bf16_t* __restrict__ out, int n)
{
  const int i = blockIdx.x * blockDim.x + threadIdx.x;
  if (i < n) out[i] = (bf16_t)in[i];
}

// ---------------------------------------------------------------------------
// Input projection: x[b,l,d] = sum_c ecg[b,c,l] * Wp[d,c] + bp[d]
// ---------------------------------------------------------------------------
__global__ __launch_bounds__(256)
void proj_in_kernel(const float* __restrict__ ecg,  // [B, 12, L]
                    const float* __restrict__ Wp,   // [512, 12]
                    const float* __restrict__ bp,   // [512]
                    float* __restrict__ x,          // [ML, 512]
                    bf16_t* __restrict__ x_bf)
{
  __shared__ float ev[NUM_LEADS];
  const int b = blockIdx.z, l = blockIdx.y;
  const int d = blockIdx.x * 256 + threadIdx.x;
  if (threadIdx.x < NUM_LEADS)
    ev[threadIdx.x] = ecg[((size_t)b * NUM_LEADS + threadIdx.x) * SEQLEN + l];
  __syncthreads();
  float acc = bp[d];
#pragma unroll
  for (int c = 0; c < NUM_LEADS; ++c) acc += ev[c] * Wp[d * NUM_LEADS + c];
  const size_t o = ((size_t)(b * SEQLEN + l)) * D_MODEL + d;
  x[o] = acc;
  x_bf[o] = (bf16_t)acc;
}

// ---------------------------------------------------------------------------
// Causal depthwise conv (D_CONV=4) + bias + SiLU over xc half of xz.
// ---------------------------------------------------------------------------
__global__ __launch_bounds__(256)
void conv_silu_kernel(const float* __restrict__ xz,     // [ML, 2048]
                      const float* __restrict__ convw,  // [1024, 4]
                      const float* __restrict__ convb,  // [1024]
                      float* __restrict__ xcc,          // [ML, 1024]
                      bf16_t* __restrict__ xc_bf)
{
  const int idx = blockIdx.x * blockDim.x + threadIdx.x;  // over ML*1024
  if (idx >= ML * D_INNER) return;
  const int e  = idx & (D_INNER - 1);
  const int bl = idx >> 10;
  const int l  = bl % SEQLEN;
  const int b  = bl / SEQLEN;
  const float* base = xz + ((size_t)b * SEQLEN) * (2 * D_INNER) + e;
  float acc = convb[e];
#pragma unroll
  for (int k = 0; k < D_CONV; ++k) {
    const int ll = l - (D_CONV - 1) + k;
    if (ll >= 0) acc += base[(size_t)ll * (2 * D_INNER)] * convw[e * D_CONV + k];
  }
  const float s = acc / (1.f + __expf(-acc));  // silu
  xcc[idx]   = s;
  xc_bf[idx] = (bf16_t)s;
}

// ---------------------------------------------------------------------------
// Selective scan, fused: dt = softplus(dbc[:, :32]·Wdtᵀ + bdt), state update,
// y = (scan_y + Dp*xc) * silu(z).  One thread per (b, e) channel.
// ---------------------------------------------------------------------------
__global__ __launch_bounds__(256)
void scan_kernel(const float* __restrict__ dbc,    // [B, L, 64]
                 const float* __restrict__ xcc,    // [B, L, 1024]
                 const float* __restrict__ xz,     // [B, L, 2048] (z at +1024)
                 const float* __restrict__ Wdt,    // [1024, 32] (layer slice)
                 const float* __restrict__ bdt,    // [1024]
                 const float* __restrict__ A_log,  // [1024, 16]
                 const float* __restrict__ Dp,     // [1024]
                 bf16_t* __restrict__ y_bf)        // [B, L, 1024]
{
  __shared__ float row[DT_RANK + 2 * D_STATE];  // 64 floats
  const int e = blockIdx.x * blockDim.x + threadIdx.x;  // 0..1023
  const int b = blockIdx.y;

  float wdt[DT_RANK];
#pragma unroll
  for (int r = 0; r < DT_RANK; ++r) wdt[r] = Wdt[e * DT_RANK + r];
  const float bdte = bdt[e];
  const float dpe  = Dp[e];
  float Av[D_STATE], h[D_STATE];
#pragma unroll
  for (int s = 0; s < D_STATE; ++s) {
    Av[s] = -__expf(A_log[e * D_STATE + s]);
    h[s]  = 0.f;
  }
  const float* dbc_b = dbc + (size_t)b * SEQLEN * 64;
  const float* xcc_b = xcc + (size_t)b * SEQLEN * D_INNER;
  const float* z_b   = xz  + (size_t)b * SEQLEN * (2 * D_INNER) + D_INNER;
  bf16_t* y_b = y_bf + (size_t)b * SEQLEN * D_INNER;

  for (int l = 0; l < SEQLEN; ++l) {
    if (threadIdx.x < 64) row[threadIdx.x] = dbc_b[(size_t)l * 64 + threadIdx.x];
    __syncthreads();
    float lin = bdte;
#pragma unroll
    for (int r = 0; r < DT_RANK; ++r) lin += row[r] * wdt[r];
    const float dt  = (lin > 20.f) ? lin : __logf(1.f + __expf(lin));
    const float x_t = xcc_b[(size_t)l * D_INNER + e];
    const float z_t = z_b[(size_t)l * (2 * D_INNER) + e];
    const float dtx = dt * x_t;
    float y = 0.f;
#pragma unroll
    for (int s = 0; s < D_STATE; ++s) {
      const float dA = __expf(dt * Av[s]);
      h[s] = h[s] * dA + dtx * row[DT_RANK + s];
      y += h[s] * row[DT_RANK + D_STATE + s];
    }
    y = (y + dpe * x_t) * (z_t / (1.f + __expf(-z_t)));
    y_b[(size_t)l * D_INNER + e] = (bf16_t)y;
    __syncthreads();
  }
}

// ---------------------------------------------------------------------------
// LayerNorm over D_MODEL=512; one block per row; also emits bf16 copy.
// ---------------------------------------------------------------------------
__global__ __launch_bounds__(256)
void layernorm_kernel(const float* __restrict__ in,  // [ML, 512]
                      const float* __restrict__ g,   // [512]
                      const float* __restrict__ bb,  // [512]
                      float* __restrict__ out,       // [ML, 512]
                      bf16_t* __restrict__ out_bf)
{
  __shared__ float red[256];
  const int tid = threadIdx.x;
  const size_t rowo = (size_t)blockIdx.x * D_MODEL;
  const float v0 = in[rowo + tid];
  const float v1 = in[rowo + tid + 256];

  red[tid] = v0 + v1;
  __syncthreads();
  for (int off = 128; off > 0; off >>= 1) {
    if (tid < off) red[tid] += red[tid + off];
    __syncthreads();
  }
  const float mean = red[0] * (1.f / D_MODEL);
  __syncthreads();
  const float d0 = v0 - mean, d1 = v1 - mean;
  red[tid] = d0 * d0 + d1 * d1;
  __syncthreads();
  for (int off = 128; off > 0; off >>= 1) {
    if (tid < off) red[tid] += red[tid + off];
    __syncthreads();
  }
  const float inv = rsqrtf(red[0] * (1.f / D_MODEL) + 1e-5f);

  const float o0 = d0 * inv * g[tid]       + bb[tid];
  const float o1 = d1 * inv * g[tid + 256] + bb[tid + 256];
  out[rowo + tid]       = o0;
  out[rowo + tid + 256] = o1;
  out_bf[rowo + tid]       = (bf16_t)o0;
  out_bf[rowo + tid + 256] = (bf16_t)o1;
}

// ---------------------------------------------------------------------------
// Host-side orchestration
// ---------------------------------------------------------------------------
static inline size_t alignup256(size_t x) { return (x + 255) & ~(size_t)255; }

extern "C" void kernel_launch(void* const* d_in, const int* in_sizes, int n_in,
                              void* d_out, int out_size, void* d_ws, size_t ws_size,
                              hipStream_t stream)
{
  (void)in_sizes; (void)n_in; (void)out_size; (void)ws_size;

  const float* ecg   = (const float*)d_in[0];
  const float* Wp    = (const float*)d_in[1];
  const float* bp    = (const float*)d_in[2];
  const float* Win   = (const float*)d_in[3];
  const float* convw = (const float*)d_in[4];
  const float* convb = (const float*)d_in[5];
  const float* Wx    = (const float*)d_in[6];
  const float* Wdt   = (const float*)d_in[7];
  const float* bdt   = (const float*)d_in[8];
  const float* A_log = (const float*)d_in[9];
  const float* Dp    = (const float*)d_in[10];
  const float* Wout  = (const float*)d_in[11];
  const float* ln_g  = (const float*)d_in[12];
  const float* ln_b  = (const float*)d_in[13];

  char* ws = (char*)d_ws;
  size_t off = 0;
  float*  x_f32 = (float*) (ws + off); off = alignup256(off + (size_t)ML * D_MODEL * 4);
  bf16_t* x_bf  = (bf16_t*)(ws + off); off = alignup256(off + (size_t)ML * D_MODEL * 2);
  float*  xz    = (float*) (ws + off); off = alignup256(off + (size_t)ML * 2 * D_INNER * 4);
  float*  xcc   = (float*) (ws + off); off = alignup256(off + (size_t)ML * D_INNER * 4);
  bf16_t* xc_bf = (bf16_t*)(ws + off); off = alignup256(off + (size_t)ML * D_INNER * 2);
  float*  dbc   = (float*) (ws + off); off = alignup256(off + (size_t)ML * 64 * 4);
  bf16_t* y_bf  = (bf16_t*)(ws + off); off = alignup256(off + (size_t)ML * D_INNER * 2);
  float*  tmp   = (float*) (ws + off); off = alignup256(off + (size_t)ML * D_MODEL * 4);
  bf16_t* Win_bf  = (bf16_t*)(ws + off); off = alignup256(off + (size_t)N_LAYERS * 2 * D_INNER * D_MODEL * 2);
  bf16_t* Wx_bf   = (bf16_t*)(ws + off); off = alignup256(off + (size_t)N_LAYERS * (DT_RANK + 2 * D_STATE) * D_INNER * 2);
  bf16_t* Wout_bf = (bf16_t*)(ws + off); off = alignup256(off + (size_t)N_LAYERS * D_MODEL * D_INNER * 2);

  // Weight conversions (deterministic, small)
  {
    int n1 = N_LAYERS * 2 * D_INNER * D_MODEL;
    int n2 = N_LAYERS * (DT_RANK + 2 * D_STATE) * D_INNER;
    int n3 = N_LAYERS * D_MODEL * D_INNER;
    f32_to_bf16_kernel<<<(n1 + 255) / 256, 256, 0, stream>>>(Win,  Win_bf,  n1);
    f32_to_bf16_kernel<<<(n2 + 255) / 256, 256, 0, stream>>>(Wx,   Wx_bf,   n2);
    f32_to_bf16_kernel<<<(n3 + 255) / 256, 256, 0, stream>>>(Wout, Wout_bf, n3);
  }

  // Input projection
  proj_in_kernel<<<dim3(D_MODEL / 256, SEQLEN, BATCH), 256, 0, stream>>>(
      ecg, Wp, bp, x_f32, x_bf);

  const dim3 gemm_block(256);
  const int mTiles = (ML + BM - 1) / BM;  // 157

  for (int l = 0; l < N_LAYERS; ++l) {
    const bf16_t* Win_l  = Win_bf  + (size_t)l * 2 * D_INNER * D_MODEL;
    const bf16_t* Wx_l   = Wx_bf   + (size_t)l * (DT_RANK + 2 * D_STATE) * D_INNER;
    const bf16_t* Wout_l = Wout_bf + (size_t)l * D_MODEL * D_INNER;

    // xz = x · Win^T   [ML, 2048]
    gemm_bf16_wmma_kernel<<<dim3(2 * D_INNER / BN, mTiles), gemm_block, 0, stream>>>(
        x_bf, Win_l, xz, ML, 2 * D_INNER, D_MODEL);

    // causal dwconv + silu on xc half
    conv_silu_kernel<<<(ML * D_INNER + 255) / 256, 256, 0, stream>>>(
        xz, convw + (size_t)l * D_INNER * D_CONV, convb + (size_t)l * D_INNER,
        xcc, xc_bf);

    // dbc = xc · Wx^T   [ML, 64]
    gemm_bf16_wmma_kernel<<<dim3(1, mTiles), gemm_block, 0, stream>>>(
        xc_bf, Wx_l, dbc, ML, DT_RANK + 2 * D_STATE, D_INNER);

    // selective scan (fused dt/softplus + gating)
    scan_kernel<<<dim3(D_INNER / 256, BATCH), 256, 0, stream>>>(
        dbc, xcc, xz,
        Wdt + (size_t)l * D_INNER * DT_RANK,
        bdt + (size_t)l * D_INNER,
        A_log + (size_t)l * D_INNER * D_STATE,
        Dp + (size_t)l * D_INNER,
        y_bf);

    // out = y · Wout^T   [ML, 512]
    gemm_bf16_wmma_kernel<<<dim3(D_MODEL / BN, mTiles), gemm_block, 0, stream>>>(
        y_bf, Wout_l, tmp, ML, D_MODEL, D_INNER);

    // layernorm -> next x (and bf16 copy); last layer writes d_out
    float* lnOut = (l == N_LAYERS - 1) ? (float*)d_out : x_f32;
    layernorm_kernel<<<ML, 256, 0, stream>>>(
        tmp, ln_g + (size_t)l * D_MODEL, ln_b + (size_t)l * D_MODEL, lnOut, x_bf);
  }
}
